// nri_decoder_32049045962803
// MI455X (gfx1250) — matrix operand order repaired and checked by reading, not compiled, not run on hardware
//
#include <hip/hip_runtime.h>

// ---------------- problem constants (from reference setup_inputs) ----------
#define BATCH 8
#define NN    1024      // nodes
#define EDGES 32768     // E = 32*N
#define FIN   128       // T*F after reshape
#define HID   256
#define FOUT  128

// ---------------- GEMM tiling ----------------------------------------------
// Block tile 128x128, TK=32. 8 waves arranged 4(M) x 2(N); each wave computes
// a 32x64 sub-tile = 2 A-frags x 4 B-frags = 8 WMMAs per k-step.
#define TM 128
#define TN 128
#define TK 32
#define AP 40           // LDS pitch (halves) for A tile rows (conflict-free)
#define BP 40           // LDS pitch (halves) for transposed B tile rows

typedef __attribute__((ext_vector_type(16))) _Float16 v16h;
typedef __attribute__((ext_vector_type(8)))  _Float16 h8;
typedef __attribute__((ext_vector_type(8)))  float    v8f;

union Frag16 { v16h v; h8 h[2]; };

// ============================================================================
// Generic batched WMMA GEMM:  C[bz] = op( A1[bz] @ B1[bz] (+ A2[bz] @ B2[bz]) )
//   A row-major f16 [M,K] (lda), B row-major f16 [K,N] (ldb), f32 accumulate.
//   Epilogue: *scale, +bias[col], optional relu; writes f32 and/or f16.
//   grid = (N/TN, M/TM, batch); all dims are exact multiples of the tiles.
// ============================================================================
__global__ __launch_bounds__(256) void k_wmma_gemm(
    const _Float16* __restrict__ A1, const _Float16* __restrict__ B1,
    const _Float16* __restrict__ A2, const _Float16* __restrict__ B2,
    float* __restrict__ C32, _Float16* __restrict__ C16,
    const float* __restrict__ bias,
    int K, int lda, int ldb, int ldc,
    long long sA, long long sB, long long sC,
    float scale, int relu)
{
    __shared__ _Float16 As[TM * AP];   // 10240 B
    __shared__ _Float16 Bs[TN * BP];   // 10240 B

    const int tid    = threadIdx.x;
    const int wave   = tid >> 5;
    const int lane   = tid & 31;
    const int half   = lane >> 4;
    const int lm     = lane & 15;
    const int wave_m = wave >> 1;      // 0..3 : 32-row strip
    const int wave_n = wave & 1;       // 0..1 : 64-col strip
    const long long m0 = (long long)blockIdx.y * TM;
    const long long n0 = (long long)blockIdx.x * TN;
    const int bz = blockIdx.z;

    const v8f vzero = {0.f,0.f,0.f,0.f,0.f,0.f,0.f,0.f};
    v8f acc[2][4];
#pragma unroll
    for (int i = 0; i < 2; ++i)
#pragma unroll
        for (int c = 0; c < 4; ++c) acc[i][c] = vzero;

    const int phases = (A2 != nullptr) ? 2 : 1;
    for (int ph = 0; ph < phases; ++ph) {
        const _Float16* __restrict__ Ag = (ph == 0 ? A1 : A2) + (long long)bz * sA;
        const _Float16* __restrict__ Bg = (ph == 0 ? B1 : B2) + (long long)bz * sB;

        for (int k0 = 0; k0 < K; k0 += TK) {
            // ---- stage A tile [TM x TK] into LDS (row-major, pitch AP) ----
#pragma unroll
            for (int t = 0; t < 2; ++t) {
                int id = tid + t * 256;          // 512 chunks of 8 halves
                int r = id >> 2, c = id & 3;
                *(h8*)&As[r * AP + c * 8] =
                    *(const h8*)&Ag[(m0 + r) * (long long)lda + k0 + c * 8];
            }
            // ---- stage B tile [TK x TN] transposed: Bs[n][k] --------------
#pragma unroll
            for (int t = 0; t < 2; ++t) {
                int id = tid + t * 256;          // 32 k-rows x 16 n-groups
                int k = id >> 4, ng = id & 15;
                h8 vb = *(const h8*)&Bg[(long long)(k0 + k) * ldb + n0 + ng * 8];
#pragma unroll
                for (int j = 0; j < 8; ++j) Bs[(ng * 8 + j) * BP + k] = vb[j];
            }
            __syncthreads();

            // ---- A fragments: rows wave_m*32 + {0,16} + lm ----------------
            Frag16 a[2];
#pragma unroll
            for (int i = 0; i < 2; ++i) {
                const int arow = (wave_m * 32 + i * 16 + lm) * AP;
                a[i].h[0] = *(const h8*)&As[arow + half * 8];
                a[i].h[1] = *(const h8*)&As[arow + 16 + half * 8];
            }
            // ---- B fragments: cols wave_n*64 + c*16 + lm ------------------
#pragma unroll
            for (int c = 0; c < 4; ++c) {
                Frag16 b;                        // column n, k = half*8+{0..7}, 16+half*8+{0..7}
                const int brow = (wave_n * 64 + c * 16 + lm) * BP;
                b.h[0] = *(const h8*)&Bs[brow + half * 8];
                b.h[1] = *(const h8*)&Bs[brow + 16 + half * 8];
                acc[0][c] = __builtin_amdgcn_wmma_f32_16x16x32_f16(
                    false, a[0].v, false, b.v, (short)0, acc[0][c], false, false);
                acc[1][c] = __builtin_amdgcn_wmma_f32_16x16x32_f16(
                    false, a[1].v, false, b.v, (short)0, acc[1][c], false, false);
            }
            __syncthreads();
        }
    }

    // ---- epilogue: C/D layout = VGPR j -> row half*8+j, col lm ------------
#pragma unroll
    for (int i = 0; i < 2; ++i) {
#pragma unroll
        for (int c = 0; c < 4; ++c) {
            const long long col = n0 + wave_n * 64 + c * 16 + lm;
            const float bv = bias ? bias[col] : 0.0f;
#pragma unroll
            for (int j = 0; j < 8; ++j) {
                const long long row = m0 + wave_m * 32 + i * 16 + half * 8 + j;
                float v = acc[i][c][j] * scale + bv;
                if (relu) v = fmaxf(v, 0.0f);
                const long long idx = (long long)bz * sC + row * (long long)ldc + col;
                if (C32) C32[idx] = v;
                if (C16) C16[idx] = (_Float16)v;
            }
        }
    }
}

// ============================================================================
// Small elementwise / graph-prep kernels
// ============================================================================
__global__ void k_fill_int(int* p, int n, int v) {
    int i = blockIdx.x * 256 + threadIdx.x;
    if (i < n) p[i] = v;
}
__global__ void k_zero_f32(float* p, long long n) {
    long long i = (long long)blockIdx.x * 256 + threadIdx.x;
    if (i < n) p[i] = 0.0f;
}
__global__ void k_cvt_f16(const float* __restrict__ in, _Float16* __restrict__ out,
                          long long n) {
    long long i = (long long)blockIdx.x * 256 + threadIdx.x;
    if (i < n) out[i] = (_Float16)in[i];
}
__global__ void k_deg_count(const int* __restrict__ dst, int* __restrict__ deg, int e) {
    int i = blockIdx.x * 256 + threadIdx.x;
    if (i < e) atomicAdd(&deg[dst[i]], 1);
}
__global__ void k_edge_norm(const int* __restrict__ src, const int* __restrict__ dst,
                            const int* __restrict__ deg, float* __restrict__ nrm, int e) {
    int i = blockIdx.x * 256 + threadIdx.x;
    if (i < e)
        nrm[i] = rsqrtf((float)deg[src[i]]) * rsqrtf((float)deg[dst[i]]);
}
// dense symmetric-normalized adjacency: adj[dst][src] += norm ; diag += 1/deg
__global__ void k_adj_edges(const int* __restrict__ src, const int* __restrict__ dst,
                            const float* __restrict__ nrm, float* __restrict__ adj, int e) {
    int i = blockIdx.x * 256 + threadIdx.x;
    if (i < e) atomicAdd(&adj[(long long)dst[i] * NN + src[i]], nrm[i]);
}
__global__ void k_adj_diag(const int* __restrict__ deg, float* __restrict__ adj, int n) {
    int i = blockIdx.x * 256 + threadIdx.x;
    if (i < n) adj[(long long)i * NN + i] += 1.0f / (float)deg[i];
}
// repack Wm[512,256] -> WmCat[256,512]: cols 0:256 = Wm[0:256,:], 256:512 = Wm[256:512,:]
__global__ void k_wmcat(const float* __restrict__ Wm, _Float16* __restrict__ out) {
    int i = blockIdx.x * 256 + threadIdx.x;   // 256*512 elements
    int k = i >> 9, n = i & 511;
    float v = (n < 256) ? Wm[k * 256 + n] : Wm[(k + 256) * 256 + (n - 256)];
    out[i] = (_Float16)v;
}
// tiled f32 [R,C] -> f16 transpose [C,R]
__global__ void k_transpose_f16(const float* __restrict__ in, _Float16* __restrict__ out,
                                int R, int C) {
    __shared__ float tile[32][33];
    int c0 = blockIdx.x * 32, r0 = blockIdx.y * 32;
    int tx = threadIdx.x, ty = threadIdx.y;   // (32, 8)
#pragma unroll
    for (int i = 0; i < 32; i += 8)
        tile[ty + i][tx] = in[(long long)(r0 + ty + i) * C + c0 + tx];
    __syncthreads();
#pragma unroll
    for (int i = 0; i < 32; i += 8)
        out[(long long)(c0 + ty + i) * R + r0 + tx] = (_Float16)tile[tx][ty + i];
}

// ============================================================================
extern "C" void kernel_launch(void* const* d_in, const int* in_sizes, int n_in,
                              void* d_out, int out_size, void* d_ws, size_t ws_size,
                              hipStream_t stream) {
    const float* x    = (const float*)d_in[0];   // [8,16,1024,8] == [8192,128]
    const int*   ei   = (const int*)d_in[1];     // [2,E] int32
    const float* m_in = (const float*)d_in[2];   // [E,N]
    const float* m_out= (const float*)d_in[3];   // [E,N]
    const float* W1   = (const float*)d_in[4];   // [128,256]
    const float* b1   = (const float*)d_in[5];
    const float* Wm   = (const float*)d_in[6];   // [512,256]
    const float* bm   = (const float*)d_in[7];
    const float* W2   = (const float*)d_in[8];   // [256,128]
    const float* b2   = (const float*)d_in[9];
    float* out = (float*)d_out;                  // [8,1024,128]
    const int* src = ei;
    const int* dst = ei + EDGES;

    // ---- workspace layout (256B aligned) ----
    size_t off = 0;
    auto take = [&](size_t bytes) -> void* {
        void* p = (char*)d_ws + off;
        off += (bytes + 255) & ~(size_t)255;
        return p;
    };
    int*      deg     = (int*)     take((size_t)NN * 4);
    float*    nrm     = (float*)   take((size_t)EDGES * 4);
    float*    adj     = (float*)   take((size_t)NN * NN * 4);
    _Float16* adj16   = (_Float16*)take((size_t)NN * NN * 2);
    _Float16* x16     = (_Float16*)take((size_t)BATCH * NN * FIN * 2);
    _Float16* W1_16   = (_Float16*)take((size_t)FIN * HID * 2);
    _Float16* WmCat16 = (_Float16*)take((size_t)HID * 2 * HID * 2);
    _Float16* W2_16   = (_Float16*)take((size_t)HID * FOUT * 2);
    _Float16* mIn16   = (_Float16*)take((size_t)EDGES * NN * 2);   // 64 MB
    _Float16* mOut16  = (_Float16*)take((size_t)EDGES * NN * 2);   // 64 MB
    _Float16* mInT16  = (_Float16*)take((size_t)NN * EDGES * 2);   // 64 MB
    _Float16* xw1_16  = (_Float16*)take((size_t)BATCH * NN * HID * 2);
    _Float16* h16     = (_Float16*)take((size_t)BATCH * NN * HID * 2);
    _Float16* G16     = (_Float16*)take((size_t)BATCH * NN * 2 * HID * 2);
    _Float16* e16     = (_Float16*)take((size_t)BATCH * EDGES * HID * 2); // 128 MB
    _Float16* h2_16   = (_Float16*)take((size_t)BATCH * NN * HID * 2);
    _Float16* xw2_16  = (_Float16*)take((size_t)BATCH * NN * FOUT * 2);
    (void)ws_size; (void)in_sizes; (void)n_in; (void)out_size;

    dim3 b256(256);
    // ---- graph preprocessing: degrees, edge norms, dense normalized adj ----
    k_fill_int <<<NN / 256, b256, 0, stream>>>(deg, NN, 1);          // self loop
    k_deg_count<<<EDGES / 256, b256, 0, stream>>>(dst, deg, EDGES);
    k_edge_norm<<<EDGES / 256, b256, 0, stream>>>(src, dst, deg, nrm, EDGES);
    k_zero_f32 <<<(NN * NN) / 256, b256, 0, stream>>>(adj, (long long)NN * NN);
    k_adj_edges<<<EDGES / 256, b256, 0, stream>>>(src, dst, nrm, adj, EDGES);
    k_adj_diag <<<NN / 256, b256, 0, stream>>>(deg, adj, NN);

    // ---- f32 -> f16 operand conversion ----
    k_cvt_f16<<<(BATCH * NN * FIN) / 256, b256, 0, stream>>>(x, x16, (long long)BATCH * NN * FIN);
    k_cvt_f16<<<(FIN * HID) / 256, b256, 0, stream>>>(W1, W1_16, (long long)FIN * HID);
    k_cvt_f16<<<(HID * FOUT) / 256, b256, 0, stream>>>(W2, W2_16, (long long)HID * FOUT);
    k_cvt_f16<<<((long long)EDGES * NN) / 256, b256, 0, stream>>>(m_in, mIn16, (long long)EDGES * NN);
    k_cvt_f16<<<((long long)EDGES * NN) / 256, b256, 0, stream>>>(m_out, mOut16, (long long)EDGES * NN);
    k_cvt_f16<<<(NN * NN) / 256, b256, 0, stream>>>(adj, adj16, (long long)NN * NN);
    k_wmcat<<<(HID * 2 * HID) / 256, b256, 0, stream>>>(Wm, WmCat16);
    k_transpose_f16<<<dim3(NN / 32, EDGES / 32), dim3(32, 8), 0, stream>>>(m_in, mInT16, EDGES, NN);

    // ---- GEMM pipeline ----
    auto gemm = [&](const _Float16* A1, const _Float16* B1,
                    const _Float16* A2, const _Float16* B2,
                    float* C32, _Float16* C16, const float* bias,
                    int M, int Nn, int K, int lda, int ldb, int ldc,
                    long long sA, long long sB, long long sC, int batch,
                    float scale, int relu) {
        dim3 g(Nn / TN, M / TM, batch);
        k_wmma_gemm<<<g, b256, 0, stream>>>(A1, B1, A2, B2, C32, C16, bias,
                                            K, lda, ldb, ldc, sA, sB, sC, scale, relu);
    };

    // 1) xw1 = x @ W1                                   [8192 x 256]
    gemm(x16, W1_16, nullptr, nullptr, nullptr, xw1_16, nullptr,
         BATCH * NN, HID, FIN, FIN, HID, HID, 0, 0, 0, 1, 1.0f, 0);
    // 2) h[b] = relu(Adj @ xw1[b] + b1)                 conv1 aggregation
    gemm(adj16, xw1_16, nullptr, nullptr, nullptr, h16, b1,
         NN, HID, NN, NN, HID, HID, 0, (long long)NN * HID, (long long)NN * HID,
         BATCH, 1.0f, 1);
    // 3) G[b] = h[b] @ [Wm_top | Wm_bot]                [1024 x 512]
    gemm(h16, WmCat16, nullptr, nullptr, nullptr, G16, nullptr,
         NN, 2 * HID, HID, HID, 2 * HID, 2 * HID,
         (long long)NN * HID, 0, (long long)NN * 2 * HID, BATCH, 1.0f, 0);
    // 4) e[b] = relu(m_in @ G[b][:, :256] + m_out @ G[b][:, 256:] + bm)  (dual-A)
    gemm(mIn16, G16, mOut16, G16 + HID, nullptr, e16, bm,
         EDGES, HID, NN, NN, 2 * HID, HID,
         0, (long long)NN * 2 * HID, (long long)EDGES * HID, BATCH, 1.0f, 1);
    // 5) h2[b] = (m_in^T @ e[b]) / N                    [1024 x 256], K = 32768
    gemm(mInT16, e16, nullptr, nullptr, nullptr, h2_16, nullptr,
         NN, HID, EDGES, EDGES, HID, HID,
         0, (long long)EDGES * HID, (long long)NN * HID, BATCH, 1.0f / NN, 0);
    // 6) xw2 = h2 @ W2                                  [8192 x 128]
    gemm(h2_16, W2_16, nullptr, nullptr, nullptr, xw2_16, nullptr,
         BATCH * NN, FOUT, HID, HID, FOUT, FOUT, 0, 0, 0, 1, 1.0f, 0);
    // 7) out[b] = Adj @ xw2[b] + b2                     conv2 aggregation -> f32
    gemm(adj16, xw2_16, nullptr, nullptr, out, nullptr, b2,
         NN, FOUT, NN, NN, FOUT, FOUT,
         0, (long long)NN * FOUT, (long long)NN * FOUT, BATCH, 1.0f, 0);
}
